// NewLLMFlexible_54434415509620
// MI455X (gfx1250) — matrix operand order repaired and checked by reading, not compiled, not run on hardware
//
#include <hip/hip_runtime.h>

#define DEVFN __device__ __forceinline__

typedef __bf16 bf16_t;
typedef bf16_t v16bf __attribute__((ext_vector_type(16)));
typedef float  v8f   __attribute__((ext_vector_type(8)));
typedef unsigned int u32x4 __attribute__((ext_vector_type(4)));
typedef int i32x4 __attribute__((ext_vector_type(4)));
typedef int i32x8 __attribute__((ext_vector_type(8)));

namespace cfg {
constexpr int V = 32000, E = 512, C = 512, H = 2048, B = 8, S = 512;
constexpr int NWG = 16;   // persistent workgroups for the recurrence
}

union Frag { v16bf v; u32x4 q[2]; };

// Optimization barrier returning an opaque zero offset (SGPR). Adding it to a
// weight pointer inside the timestep loop stops LICM from hoisting (and then
// spilling) the fragment loads, WITHOUT breaking the pointer's global
// address-space provenance -- loads stay global_load_b128, not flat_load.
DEVFN long long opaque_off() {
  long long x = 0;
  asm volatile("" : "+s"(x));
  return x;
}

// A-fragment (16x32 bf16, M x K) from a row-major [16][ld] buffer.
DEVFN v16bf load_afrag(const bf16_t* A, int ld, int kt, int lane) {
  int row = lane & 15;
  int sel = (lane >> 4) & 1;
  const bf16_t* p = A + (long long)row * ld + kt * 32 + sel * 8;
  Frag f;
  f.q[0] = *(const u32x4*)(p);
  f.q[1] = *(const u32x4*)(p + 16);
  return f.v;
}

// B-fragment (32x16 bf16, K x N) from pre-packed tiles (32 contiguous B/lane).
DEVFN v16bf load_bfrag(const bf16_t* Wp, int ktiles, int kt, int nt, int lane) {
  const bf16_t* p = Wp + (((long long)nt * ktiles + kt) * 32 + lane) * 16;
  Frag f;
  f.q[0] = *(const u32x4*)(p);
  f.q[1] = *(const u32x4*)(p + 8);
  return f.v;
}

DEVFN v8f wmma_bf16(v16bf a, v16bf b, v8f c) {
  return __builtin_amdgcn_wmma_f32_16x16x32_bf16(false, a, false, b, (short)0, c,
                                                 false, false);
}

// Software-pipelined K-loop, prefetch distance 4: fragments for kt+4 are
// issued before the WMMA on kt (s_wait_loadcnt <= 12, ~4 iterations of L2
// latency in flight). Fully unrolled so the rotating slots become registers.
DEVFN v8f gemm_pipe(const bf16_t* __restrict__ A, int ld, int ktA0,
                    const bf16_t* __restrict__ Wp, int ktiles, int ktB0,
                    int nkt, int nt, int lane, v8f acc) {
  constexpr int D = 4;
  v16bf af[D], bf_[D];
#pragma unroll
  for (int i = 0; i < D; ++i) {
    if (i < nkt) {
      af[i]  = load_afrag(A, ld, ktA0 + i, lane);
      bf_[i] = load_bfrag(Wp, ktiles, ktB0 + i, nt, lane);
    }
  }
#pragma unroll
  for (int k = 0; k < nkt; ++k) {
    v16bf a = af[k % D], b = bf_[k % D];
    if (k + D < nkt) {  // issue the k+D loads before waiting on k's
      af[k % D]  = load_afrag(A, ld, ktA0 + k + D, lane);
      bf_[k % D] = load_bfrag(Wp, ktiles, ktB0 + k + D, nt, lane);
    }
    acc = wmma_bf16(a, b, acc);
  }
  return acc;
}

DEVFN float sigmoidf_(float x) { return 1.f / (1.f + __expf(-x)); }

// ---------------------------------------------------------------------------
// Tensor Data Mover: 2D tile (tile_d0 x tile_d1 elements, 2-byte data) from a
// row-major tensor (tensor_d0 x tensor_d1, stride_elems) into LDS at lds_off.
// D# layout per CDNA5 ISA sect 8.3/8.4.
// ---------------------------------------------------------------------------
#if __has_builtin(__builtin_amdgcn_tensor_load_to_lds)
#define HAVE_TDM 1
#else
#define HAVE_TDM 0
#endif

#if HAVE_TDM
DEVFN void tdm_load_2d(unsigned lds_off, const void* gaddr,
                       unsigned tensor_d0, unsigned tensor_d1,
                       unsigned tile_d0, unsigned tile_d1,
                       unsigned stride_elems) {
  unsigned long long ga = (unsigned long long)gaddr;
  u32x4 g0;
  g0[0] = 1u;                                   // count=1, user descriptor
  g0[1] = lds_off;                              // lds_addr (bytes)
  g0[2] = (unsigned)(ga & 0xffffffffu);         // global_addr[31:0]
  g0[3] = (unsigned)(ga >> 32) | (2u << 30);    // global_addr[56:32] | type=2
  i32x8 g1;
  g1[0] = (int)(1u << 16);                      // data_size=1 (2 bytes/elem)
  g1[1] = (int)((tensor_d0 & 0xffffu) << 16);   // tensor_dim0[15:0] @ bits 63:48
  g1[2] = (int)(((tensor_d0 >> 16) & 0xffffu) | ((tensor_d1 & 0xffffu) << 16));
  g1[3] = (int)(((tensor_d1 >> 16) & 0xffffu) | ((tile_d0 & 0xffffu) << 16));
  g1[4] = (int)(tile_d1 & 0xffffu);             // tile_dim1 (tile_dim2 = 0)
  g1[5] = (int)stride_elems;                    // tensor_dim0_stride[31:0]
  g1[6] = 0;
  g1[7] = 0;
  i32x4 z4 = {0, 0, 0, 0};
#if __clang_major__ >= 23
  i32x8 z8 = {0, 0, 0, 0, 0, 0, 0, 0};
  __builtin_amdgcn_tensor_load_to_lds(g0, g1, z4, z4, z8, 0);
#else
  __builtin_amdgcn_tensor_load_to_lds(g0, g1, z4, z4, 0);
#endif
}
#endif

// ---------------------------------------------------------------------------
// Phase 0: repack fp32 weights [K][N] row-major -> bf16 B-fragment tiles.
// ---------------------------------------------------------------------------
__global__ void repack_b_kernel(const float* __restrict__ W, bf16_t* __restrict__ out,
                                int K, int N, long long total) {
  long long tid = (long long)blockIdx.x * 256 + threadIdx.x;
  if (tid >= total) return;
  int e = (int)(tid & 15);
  int lane = (int)((tid >> 4) & 31);
  long long tile = tid >> 9;
  int ktiles = K >> 5;
  int kt = (int)(tile % ktiles);
  int nt = (int)(tile / ktiles);
  int k = kt * 32 + ((lane >= 16) ? 8 : 0) + e + ((e >= 8) ? 8 : 0);
  int n = nt * 16 + (lane & 15);
  out[tid] = (bf16_t)W[(long long)k * N + n];
}

// ---------------------------------------------------------------------------
// Phase 1: token embedding + LayerNorm -> bf16 te[S][16][E] (rows 8..15 = 0)
// ---------------------------------------------------------------------------
__global__ void embed_ln_kernel(const int* __restrict__ ids, const float* __restrict__ emb,
                                const float* __restrict__ g, const float* __restrict__ bta,
                                bf16_t* __restrict__ te) {
  using namespace cfg;
  int bs = blockIdx.x;
  int b = bs / S, s = bs % S;
  int id = ids[bs];
  const float* row = emb + (long long)id * E;
  int t = threadIdx.x, lane = t & 31, wv = t >> 5;

  float x0 = row[t], x1 = row[t + 256];
  float sum = x0 + x1, sq = x0 * x0 + x1 * x1;
  for (int off = 16; off >= 1; off >>= 1) {
    sum += __shfl_xor(sum, off, 32);
    sq  += __shfl_xor(sq,  off, 32);
  }
  __shared__ float ssum[8], ssq[8];
  if (lane == 0) { ssum[wv] = sum; ssq[wv] = sq; }
  __syncthreads();
  if (wv == 0) {
    float a = (lane < 8) ? ssum[lane] : 0.f;
    float c = (lane < 8) ? ssq[lane]  : 0.f;
    for (int off = 4; off >= 1; off >>= 1) {
      a += __shfl_xor(a, off, 32);
      c += __shfl_xor(c, off, 32);
    }
    if (lane == 0) { ssum[0] = a; ssq[0] = c; }
  }
  __syncthreads();
  float mean = ssum[0] * (1.f / E);
  float var = ssq[0] * (1.f / E) - mean * mean;
  float rstd = rsqrtf(var + 1e-5f);
  bf16_t* o = te + ((long long)s * 16 + b) * E;
  o[t]       = (bf16_t)((x0 - mean) * rstd * g[t]       + bta[t]);
  o[t + 256] = (bf16_t)((x1 - mean) * rstd * g[t + 256] + bta[t + 256]);
}

// ---------------------------------------------------------------------------
// Grid-wide barrier for the persistent recurrence kernel (16 co-resident WGs).
// ---------------------------------------------------------------------------
DEVFN void grid_sync(unsigned* bar, unsigned* phase, int nwg) {
  __syncthreads();
  if (threadIdx.x == 0) {
    __threadfence();                       // release
    unsigned target = (++(*phase)) * (unsigned)nwg;
    atomicAdd(bar, 1u);
    while (__hip_atomic_load(bar, __ATOMIC_RELAXED, __HIP_MEMORY_SCOPE_AGENT) < target)
      __builtin_amdgcn_s_sleep(2);
    __threadfence();                       // acquire
  }
  __syncthreads();
}

// ---------------------------------------------------------------------------
// Phase 2: sequential recurrence. 16 WGs x 8 waves = 128 waves = H/16 tiles.
// ---------------------------------------------------------------------------
__global__ void __launch_bounds__(256)
recurrence_kernel(const bf16_t* __restrict__ te, const bf16_t* __restrict__ W0P,
                  const bf16_t* __restrict__ W1P, const bf16_t* __restrict__ WdP,
                  const bf16_t* __restrict__ WfP, const bf16_t* __restrict__ WiP,
                  const float* __restrict__ b0, const float* __restrict__ b1,
                  const float* __restrict__ bd, const float* __restrict__ bfv,
                  const float* __restrict__ bi, const float* __restrict__ cg,
                  const float* __restrict__ cb,
                  bf16_t* __restrict__ h1A, bf16_t* __restrict__ h2A,
                  float* __restrict__ dfi, bf16_t* __restrict__ hP,
                  unsigned* __restrict__ bar) {
  using namespace cfg;
  __shared__ __attribute__((aligned(16))) bf16_t ctxA[16 * C];  // LN'd ctx, A-layout
  __shared__ __attribute__((aligned(16))) float  ctxF[8 * C];   // fp32 copy

  int tid = threadIdx.x, lane = tid & 31, wv = tid >> 5;
  int gw = blockIdx.x * 8 + wv;  // 0..127

  for (int i = tid; i < 16 * C; i += 256) ctxA[i] = (bf16_t)0.f;
  for (int i = tid; i < 8 * C; i += 256) ctxF[i] = 0.f;
  __syncthreads();

  unsigned phase = 0;
  for (int s = 0; s < S; ++s) {
    const bf16_t* teS = te + (long long)s * 16 * E;
    // Defeat LICM (see opaque_off): weight fragment loads are re-read from
    // the L2-resident pack each step instead of being hoisted and spilled.
    long long zo = opaque_off();
    const bf16_t* W0p = W0P + zo;
    const bf16_t* W1p = W1P + zo;
    const bf16_t* Wdp = WdP + zo;
    const bf16_t* Wfp = WfP + zo;
    const bf16_t* Wip = WiP + zo;

    // -------- h1 = relu([x, ctx] @ W0 + b0) --------
    {
      int nt = gw;
      int col = nt * 16 + (lane & 15);
      v8f acc = {};
      acc = gemm_pipe(teS,  E, 0, W0p, 32, 0,  16, nt, lane, acc);  // x part
      acc = gemm_pipe(ctxA, C, 0, W0p, 32, 16, 16, nt, lane, acc);  // ctx (LDS)
      float bv = b0[col];
      int mo = (lane >= 16) ? 8 : 0;
#pragma unroll
      for (int r = 0; r < 8; ++r)
        h1A[(r + mo) * H + col] = (bf16_t)fmaxf(acc[r] + bv, 0.f);
    }
    grid_sync(bar, &phase, NWG);

    // -------- h2 = relu(h1 @ W1 + b1) --------
    {
      int nt = gw;
      int col = nt * 16 + (lane & 15);
      v8f acc = {};
      acc = gemm_pipe(h1A, H, 0, W1p, 64, 0, 64, nt, lane, acc);
      float bv = b1[col];
      int mo = (lane >= 16) ? 8 : 0;
#pragma unroll
      for (int r = 0; r < 8; ++r)
        h2A[(r + mo) * H + col] = (bf16_t)fmaxf(acc[r] + bv, 0.f);
      if (lane < 16) {  // valid batch rows -> two steps per 16-row M-tile
        long long row0 = (long long)(s >> 1) * 16 + (s & 1) * 8;
#pragma unroll
        for (int r = 0; r < 8; ++r)
          hP[(row0 + r) * H + col] = (bf16_t)fmaxf(acc[r] + bv, 0.f);
      }
    }
    grid_sync(bar, &phase, NWG);

    // -------- d/f/i pre-activations (waves 0..95) --------
    if (gw < 96) {
      int mat = gw >> 5, nt = gw & 31;
      int col = nt * 16 + (lane & 15);
      const bf16_t* Wp = (mat == 0) ? Wdp : ((mat == 1) ? Wfp : Wip);
      const float* bias = (mat == 0) ? bd : ((mat == 1) ? bfv : bi);
      v8f acc = {};
      acc = gemm_pipe(h2A, H, 0, Wp, 64, 0, 64, nt, lane, acc);
      float bv = bias[col];
      int mo = (lane >= 16) ? 8 : 0;
#pragma unroll
      for (int r = 0; r < 8; ++r)
        dfi[mat * 16 * C + (r + mo) * C + col] = acc[r] + bv;
    }
    grid_sync(bar, &phase, NWG);

    // -------- ctx = clip(LN(f*ctx + i*tanh(d))), redundant per WG ----------
    if (wv < 8) {
      float nc[16];
      float sum = 0.f, sq = 0.f;
#pragma unroll
      for (int j = 0; j < 16; ++j) {
        int c = lane + j * 32;
        float dr = dfi[0 * 16 * C + wv * C + c];
        float fr = dfi[1 * 16 * C + wv * C + c];
        float ir = dfi[2 * 16 * C + wv * C + c];
        float co = ctxF[wv * C + c];
        float v = sigmoidf_(fr) * co + sigmoidf_(ir) * tanhf(dr);
        nc[j] = v;
        sum += v;
        sq += v * v;
      }
      for (int off = 16; off >= 1; off >>= 1) {
        sum += __shfl_xor(sum, off, 32);
        sq  += __shfl_xor(sq,  off, 32);
      }
      float mean = sum * (1.f / C);
      float var = sq * (1.f / C) - mean * mean;
      float rstd = rsqrtf(var + 1e-5f);
#pragma unroll
      for (int j = 0; j < 16; ++j) {
        int c = lane + j * 32;
        float y = (nc[j] - mean) * rstd * cg[c] + cb[c];
        y = fminf(fmaxf(y, -10.f), 10.f);
        ctxF[wv * C + c] = y;
        ctxA[wv * C + c] = (bf16_t)y;
      }
    }
    __syncthreads();
  }
}

// ---------------------------------------------------------------------------
// Phase 3: logits = hPack[4096,2048] @ Wout[2048,32000] + bout  (537 GFLOP)
// Block tile 64x128. A chunks (64x512 bf16 = 64 KB) staged by the Tensor Data
// Mover into double-buffered LDS, overlapped with WMMA; B fragments from the
// L2-resident bf16 Wout, pipelined and reused across 4 M-tiles per wave.
// ---------------------------------------------------------------------------
__global__ void __launch_bounds__(256)
logits_kernel(const bf16_t* __restrict__ hP, const bf16_t* __restrict__ WoutP,
              const float* __restrict__ bout, float* __restrict__ out) {
  using namespace cfg;
  __shared__ __attribute__((aligned(16))) bf16_t As[2][64 * 512];  // 2 x 64 KB
  int tid = threadIdx.x, lane = tid & 31, wv = tid >> 5;
  long long r0 = (long long)blockIdx.y * 64;
  int ntile = blockIdx.x * 8 + wv;
  const bf16_t* gbase = hP + r0 * H;

#if HAVE_TDM
  unsigned ldsOff[2] = {(unsigned)(size_t)(void*)&As[0][0],
                        (unsigned)(size_t)(void*)&As[1][0]};
  if (wv == 0)  // one DMA per workgroup; EXEC-independent, tracked by TENSORcnt
    tdm_load_2d(ldsOff[0], gbase, /*tensor_d0=*/H, /*tensor_d1=*/B * S,
                /*tile_d0=*/512, /*tile_d1=*/64, /*stride=*/H);
#endif

  v8f acc[4] = {};
#pragma unroll
  for (int kc = 0; kc < 4; ++kc) {
#if HAVE_TDM
    if (wv == 0) {
      if (kc + 1 < 4) {
        tdm_load_2d(ldsOff[(kc + 1) & 1], gbase + (kc + 1) * 512,
                    H, B * S, 512, 64, H);
        __builtin_amdgcn_s_wait_tensorcnt(1);  // chunk kc landed; kc+1 in flight
      } else {
        __builtin_amdgcn_s_wait_tensorcnt(0);  // last chunk landed
      }
    }
    __syncthreads();  // LDS chunk kc visible to all waves
#else
    __syncthreads();
    for (int i = tid; i < 64 * 64; i += 256) {  // fallback: cooperative stage
      int row = i >> 6, u = i & 63;
      u32x4 q = *(const u32x4*)(gbase + row * H + kc * 512 + u * 8);
      *(u32x4*)(&As[kc & 1][row * 512 + u * 8]) = q;
    }
    __syncthreads();
#endif
    const bf16_t* Ab = &As[kc & 1][0];
    // B-pipelined compute: 16 K-steps, B reused by 4 M-tiles
    v16bf bb0 = load_bfrag(WoutP, 64, kc * 16, ntile, lane);
#pragma unroll
    for (int ks = 0; ks < 16; ++ks) {
      int kt = kc * 16 + ks;
      v16bf bb1 = bb0;
      if (ks + 1 < 16)
        bb1 = load_bfrag(WoutP, 64, kt + 1, ntile, lane);
      if (kt + 2 < 64) {  // keep the B stream hot ahead of the pipeline
        const bf16_t* np = WoutP + (((long long)ntile * 64 + kt + 2) * 32 + lane) * 16;
        __builtin_prefetch(np, 0, 1);
      }
#pragma unroll
      for (int mt = 0; mt < 4; ++mt)
        acc[mt] = wmma_bf16(load_afrag(Ab + mt * 16 * 512, 512, ks, lane), bb0, acc[mt]);
      bb0 = bb1;
    }
    __syncthreads();  // all waves done with chunk kc before DMA overwrites it
  }

  int v = ntile * 16 + (lane & 15);
  float bv = bout[v];
  int mo = (lane >= 16) ? 8 : 0;
#pragma unroll
  for (int mt = 0; mt < 4; ++mt) {
    int Mt = blockIdx.y * 4 + mt;
#pragma unroll
    for (int r = 0; r < 8; ++r) {
      int m = r + mo;
      int s = 2 * Mt + (m >> 3);
      int b = m & 7;
      out[((long long)b * S + s) * V + v] = acc[mt][r] + bv;
    }
  }
}

// ---------------------------------------------------------------------------
extern "C" void kernel_launch(void* const* d_in, const int* in_sizes, int n_in,
                              void* d_out, int out_size, void* d_ws, size_t ws_size,
                              hipStream_t stream) {
  using namespace cfg;
  (void)in_sizes; (void)n_in; (void)out_size; (void)ws_size;
  const int*   ids  = (const int*)d_in[0];
  const float* emb  = (const float*)d_in[1];
  const float* lneg = (const float*)d_in[2];
  const float* lneb = (const float*)d_in[3];
  const float* W0   = (const float*)d_in[4];
  const float* b0   = (const float*)d_in[5];
  const float* W1   = (const float*)d_in[6];
  const float* b1   = (const float*)d_in[7];
  const float* Wout = (const float*)d_in[8];
  const float* bout = (const float*)d_in[9];
  const float* Wd   = (const float*)d_in[10];
  const float* bd   = (const float*)d_in[11];
  const float* Wf   = (const float*)d_in[12];
  const float* bfp  = (const float*)d_in[13];
  const float* Wi   = (const float*)d_in[14];
  const float* bi   = (const float*)d_in[15];
  const float* lncg = (const float*)d_in[16];
  const float* lncb = (const float*)d_in[17];
  float* out = (float*)d_out;

  char* base = (char*)d_ws;
  size_t off = 0;
  auto take = [&](size_t bytes) -> char* {
    char* p = base + off;
    off += (bytes + 255) & ~(size_t)255;
    return p;
  };
  unsigned* bar  = (unsigned*)take(256);
  bf16_t* tePad  = (bf16_t*)take((size_t)S * 16 * E * 2);
  bf16_t* W0P    = (bf16_t*)take((size_t)(E + C) * H * 2);
  bf16_t* W1P    = (bf16_t*)take((size_t)H * H * 2);
  bf16_t* WdP    = (bf16_t*)take((size_t)H * C * 2);
  bf16_t* WfP    = (bf16_t*)take((size_t)H * C * 2);
  bf16_t* WiP    = (bf16_t*)take((size_t)H * C * 2);
  bf16_t* WoutP  = (bf16_t*)take((size_t)H * V * 2);
  bf16_t* h1A    = (bf16_t*)take((size_t)16 * H * 2);
  bf16_t* h2A    = (bf16_t*)take((size_t)16 * H * 2);
  float*  dfi    = (float*)take((size_t)3 * 16 * C * 4);
  bf16_t* hPack  = (bf16_t*)take((size_t)B * S * H * 2);

  (void)hipMemsetAsync(bar, 0, 256, stream);
  (void)hipMemsetAsync(tePad, 0, (size_t)S * 16 * E * 2, stream);  // zero pad rows

  auto repack = [&](const float* W, bf16_t* o, int K, int N) {
    long long total = (long long)K * N;
    int blocks = (int)((total + 255) / 256);
    repack_b_kernel<<<blocks, 256, 0, stream>>>(W, o, K, N, total);
  };
  repack(W0, W0P, E + C, H);
  repack(W1, W1P, H, H);
  repack(Wd, WdP, H, C);
  repack(Wf, WfP, H, C);
  repack(Wi, WiP, H, C);
  repack(Wout, WoutP, H, V);

  embed_ln_kernel<<<B * S, 256, 0, stream>>>(ids, emb, lneg, lneb, tePad);

  recurrence_kernel<<<NWG, 256, 0, stream>>>(tePad, W0P, W1P, WdP, WfP, WiP,
                                             b0, b1, bd, bfp, bi, lncg, lncb,
                                             h1A, h2A, dfi, hPack, bar);

  logits_kernel<<<dim3(V / 16 / 8, (B * S) / 64), 256, 0, stream>>>(hPack, WoutP,
                                                                    bout, out);
}
